// DifferentialTransformerEncoder_51092930953784
// MI455X (gfx1250) — matrix operand order repaired and checked by reading, not compiled, not run on hardware
//
#include <hip/hip_runtime.h>
#include <math.h>
#include <stdint.h>

// ---------------------------------------------------------------------------
// MI455X (gfx1250) differential transformer encoder.
//  * All GEMMs + attention matmuls: v_wmma_f32_16x16x32_bf16 (f32 accumulate).
//  * Weights pre-converted once per launch to bf16, transposed [N][K], so GEMM
//    B-tiles stream via global_load_async_to_lds_b128 (ASYNCcnt) with no VALU.
//  * Fragment gathers use 16-byte LDS/global vector loads per the CDNA5
//    16-bit A/B VGPR layouts (two contiguous 8-element runs per fragment).
// ---------------------------------------------------------------------------

typedef __attribute__((ext_vector_type(16))) __bf16 v16bf;
typedef __attribute__((ext_vector_type(8)))  __bf16 v8bf;
typedef __attribute__((ext_vector_type(8)))  float  v8f;

#define BDIM 1024   // D
#define SEQL 1024   // L
#define NBAT 4      // B
#define NHEAD 16    // H
#define HDIM 64     // HD
#define DFFN 4096   // DFF

#define ACT_NONE    0
#define ACT_TANH    1
#define ACT_GELU    2
#define ACT_SIGMOID 3

__device__ __forceinline__ __bf16 to_bf(float f) { return (__bf16)f; }

__device__ __forceinline__ v8f v8zero() {
  v8f z = {0.f, 0.f, 0.f, 0.f, 0.f, 0.f, 0.f, 0.f};
  return z;
}

__device__ __forceinline__ v8f wmma_bf16(v16bf a, v16bf b, v8f c) {
  return __builtin_amdgcn_wmma_f32_16x16x32_bf16(false, a, false, b,
                                                 (short)0, c, false, false);
}

__device__ __forceinline__ v16bf cat8(v8bf lo, v8bf hi) {
  return __builtin_shufflevector(lo, hi, 0, 1, 2, 3, 4, 5, 6, 7,
                                 8, 9, 10, 11, 12, 13, 14, 15);
}

// Async global->LDS copy (16B per lane), tracked by ASYNCcnt.
__device__ __forceinline__ void async_load_b128(unsigned lds_off,
                                                const void* gptr) {
  asm volatile("global_load_async_to_lds_b128 %0, %1, off"
               :
               : "v"(lds_off), "v"((unsigned long long)(size_t)gptr)
               : "memory");
}
__device__ __forceinline__ void wait_async0() {
  asm volatile("s_wait_asynccnt 0x0" ::: "memory");
}

// A-matrix (16x32, 16-bit): lane half kh selects +8 within each 16-K block;
// elements 0..7 live at K = kh*8 + 0..7, elements 8..15 at K = 16 + kh*8 + 0..7.
__device__ __forceinline__ v16bf lds_frag_a(const __bf16* buf, int stride,
                                            int rowBase, int kBase, int lane) {
  const int r  = rowBase + (lane & 15);
  const int kh = (lane >> 4) & 1;
  const __bf16* p = buf + r * stride + kBase + kh * 8;
  const v8bf lo = *reinterpret_cast<const v8bf*>(p);
  const v8bf hi = *reinterpret_cast<const v8bf*>(p + 16);
  return cat8(lo, hi);
}

// B-matrix (32x16, 16-bit): same K order as A: elements 0..7 at K = kh*8 + 0..7, 8..15 at K = 16 + kh*8 + 0..7.
__device__ __forceinline__ v16bf lds_frag_b(const __bf16* buf, int stride,
                                            int nBase, int kBase, int lane) {
  const int n  = nBase + (lane & 15);
  const int kh = (lane >> 4) & 1;
  const __bf16* p = buf + n * stride + kBase + kh * 8;
  const v8bf lo = *reinterpret_cast<const v8bf*>(p);
  const v8bf hi = *reinterpret_cast<const v8bf*>(p + 16);
  return cat8(lo, hi);
}

// Block-wide sum over 256 threads.
__device__ __forceinline__ float block_sum(float v) {
  __shared__ float sred[256];
  const int tid = threadIdx.x;
  sred[tid] = v;
  __syncthreads();
#pragma unroll
  for (int s = 128; s > 0; s >>= 1) {
    if (tid < s) sred[tid] += sred[tid + s];
    __syncthreads();
  }
  const float r = sred[0];
  __syncthreads();
  return r;
}

// ---------------------------------------------------------------------------
// Weight prep: W (K x N, f32, row-major) -> Wt (N x K, bf16). 32x32 LDS tiles.
// ---------------------------------------------------------------------------
__global__ __launch_bounds__(256)
void wtrans_kernel(const float* __restrict__ W, __bf16* __restrict__ Wt,
                   int K, int N) {
  __shared__ float tile[32][33];
  const int n0 = blockIdx.x * 32;
  const int k0 = blockIdx.y * 32;
  const int tx = threadIdx.x;   // 0..31
  const int ty = threadIdx.y;   // 0..7
#pragma unroll
  for (int j = 0; j < 32; j += 8)
    tile[ty + j][tx] = W[(size_t)(k0 + ty + j) * N + n0 + tx];
  __syncthreads();
#pragma unroll
  for (int j = 0; j < 32; j += 8)
    Wt[(size_t)(n0 + ty + j) * K + k0 + tx] = to_bf(tile[tx][ty + j]);
}

// ---------------------------------------------------------------------------
// GEMM: C(MxN,f32) = act(A(MxK,f32) @ W + bias), W given as Wt (NxK, bf16).
// Block tile 128x128, BK=32, 256 threads = 8 waves, wave tile 64x32.
// A staged sync (f32->bf16); B staged via async b128 copies (no conversion).
// ---------------------------------------------------------------------------
template <int ACT>
__global__ __launch_bounds__(256)
void gemm_bf16(const float* __restrict__ A, const __bf16* __restrict__ Bt,
               const float* __restrict__ bias, float* __restrict__ C,
               int M, int N, int K) {
  __shared__ __bf16 As[128 * 40];  // [m][k], stride 40 (pad 8)
  __shared__ __bf16 Bs[128 * 40];  // [n][k], stride 40

  const int tid    = threadIdx.x;
  const int lane   = tid & 31;
  const int wid    = tid >> 5;
  const int wm     = (wid >> 2) * 64;
  const int wn     = (wid & 3) * 32;
  const int mBlock = blockIdx.y * 128;
  const int nBlock = blockIdx.x * 128;

  v8f acc[4][2];
#pragma unroll
  for (int mi = 0; mi < 4; ++mi)
#pragma unroll
    for (int ni = 0; ni < 2; ++ni) acc[mi][ni] = v8zero();

  for (int kt = 0; kt < K; kt += 32) {
    // Async-stage B tile: 128 rows x 32 k of bf16 = 512 x 16B transfers.
#pragma unroll
    for (int it = 0; it < 2; ++it) {
      const int f   = tid + it * 256;     // 0..511
      const int n   = f >> 2;
      const int sub = (f & 3) * 8;        // bf16 elements
      const __bf16* src = Bt + (size_t)(nBlock + n) * K + kt + sub;
      const unsigned lds = (unsigned)(size_t)(const void*)&Bs[n * 40 + sub];
      async_load_b128(lds, src);
    }
    // Sync-stage A tile (128x32 f32 -> bf16).
#pragma unroll
    for (int it = 0; it < 4; ++it) {
      const int f = tid + it * 256;       // 1024 float4 slots
      const int r = f >> 3;
      const int c = (f & 7) << 2;
      const float4 a4 =
          *reinterpret_cast<const float4*>(A + (size_t)(mBlock + r) * K + kt + c);
      __bf16* dst = &As[r * 40 + c];
      dst[0] = to_bf(a4.x); dst[1] = to_bf(a4.y);
      dst[2] = to_bf(a4.z); dst[3] = to_bf(a4.w);
    }
    wait_async0();
    __syncthreads();

    const v16bf bf0 = lds_frag_b(Bs, 40, wn +  0, 0, lane);
    const v16bf bf1 = lds_frag_b(Bs, 40, wn + 16, 0, lane);
#pragma unroll
    for (int mi = 0; mi < 4; ++mi) {
      const v16bf af = lds_frag_a(As, 40, wm + mi * 16, 0, lane);
      acc[mi][0] = wmma_bf16(af, bf0, acc[mi][0]);
      acc[mi][1] = wmma_bf16(af, bf1, acc[mi][1]);
    }
    __syncthreads();
  }

  // Epilogue. C/D layout: lane holds n = lane&15, VGPR r -> m = r + 8*(lane>>4).
#pragma unroll
  for (int mi = 0; mi < 4; ++mi) {
#pragma unroll
    for (int ni = 0; ni < 2; ++ni) {
      const int col = nBlock + wn + ni * 16 + (lane & 15);
      const float bv = bias ? bias[col] : 0.f;
#pragma unroll
      for (int r = 0; r < 8; ++r) {
        const int row = mBlock + wm + mi * 16 + r + ((lane >> 4) << 3);
        float v = acc[mi][ni][r] + bv;
        if (ACT == ACT_TANH) {
          v = tanhf(v);
        } else if (ACT == ACT_GELU) {
          v = 0.5f * v * (1.f + tanhf(0.7978845608f * (v + 0.044715f * v * v * v)));
        } else if (ACT == ACT_SIGMOID) {
          v = 1.f / (1.f + expf(-v));
        }
        C[(size_t)row * N + col] = v;
      }
    }
  }
}

// ---------------------------------------------------------------------------
// Differential attention for one (b,h,16-query-row) block.
// qkv layout: row (b*L + l), 3072 cols = [s(0..2)][h][hd].
// Uses s2[i][j] = s1[i][j-1 mod L] (rolling K == rolling score columns).
// LDS: raw scores sS[16][1024] bf16 + diff sD[16][1024] bf16 = 64 KB.
// ---------------------------------------------------------------------------
__global__ __launch_bounds__(256)
void attn_kernel(const float* __restrict__ qkv, const int* __restrict__ mask,
                 const float* __restrict__ lam, int layer,
                 float* __restrict__ attn) {
  __shared__ __bf16 sS[16 * 1024];
  __shared__ __bf16 sD[16 * 1024];

  const int tid  = threadIdx.x;
  const int lane = tid & 31;
  const int wid  = tid >> 5;
  const int bh   = blockIdx.y;
  const int b    = bh >> 4;
  const int h    = bh & 15;
  const int rowBase = blockIdx.x * 16;
  const size_t rstride = 3 * BDIM;
  const float scale = 0.03125f;        // D**-0.5 (reference scales by D)

  // Q fragments (16x64 -> two K=32 steps), A layout, vector global loads.
  v16bf qf[2];
  {
    const int m  = rowBase + (lane & 15);
    const int kh = (lane >> 4) & 1;
    const float* qrow = qkv + (size_t)(b * SEQL + m) * rstride + h * HDIM;
#pragma unroll
    for (int ks = 0; ks < 2; ++ks) {
      const float* g0 = qrow + ks * 32 + kh * 8;        // elements 0..7
      const float* g1 = qrow + ks * 32 + 16 + kh * 8;   // elements 8..15
      const float4 a0 = *reinterpret_cast<const float4*>(g0);
      const float4 a1 = *reinterpret_cast<const float4*>(g0 + 4);
      const float4 b0 = *reinterpret_cast<const float4*>(g1);
      const float4 b1 = *reinterpret_cast<const float4*>(g1 + 4);
      v16bf f;
      f[0] = to_bf(a0.x); f[1] = to_bf(a0.y); f[2]  = to_bf(a0.z); f[3]  = to_bf(a0.w);
      f[4] = to_bf(a1.x); f[5] = to_bf(a1.y); f[6]  = to_bf(a1.z); f[7]  = to_bf(a1.w);
      f[8] = to_bf(b0.x); f[9] = to_bf(b0.y); f[10] = to_bf(b0.z); f[11] = to_bf(b0.w);
      f[12] = to_bf(b1.x); f[13] = to_bf(b1.y); f[14] = to_bf(b1.z); f[15] = to_bf(b1.w);
      qf[ks] = f;
    }
  }

  // Phase 1: raw scores s[i][j] = scale * (q_i . k_j) -> bf16 LDS.
  {
    const int n  = lane & 15;
    const int kh = (lane >> 4) & 1;
    for (int j0 = 0; j0 < SEQL; j0 += 128) {
      const int jc = j0 + wid * 16;
      const float* krow =
          qkv + (size_t)(b * SEQL + jc + n) * rstride + BDIM + h * HDIM;
      v8f acc = v8zero();
#pragma unroll
      for (int ks = 0; ks < 2; ++ks) {
        // B layout: elements 0..7 at ks*32 + kh*8, elements 8..15 at ks*32 + 16 + kh*8.
        const float* g = krow + ks * 32 + kh * 8;
        const float4 c0 = *reinterpret_cast<const float4*>(g);
        const float4 c1 = *reinterpret_cast<const float4*>(g + 4);
        const float4 c2 = *reinterpret_cast<const float4*>(g + 16);
        const float4 c3 = *reinterpret_cast<const float4*>(g + 20);
        v16bf kf;
        kf[0] = to_bf(c0.x); kf[1] = to_bf(c0.y); kf[2]  = to_bf(c0.z); kf[3]  = to_bf(c0.w);
        kf[4] = to_bf(c1.x); kf[5] = to_bf(c1.y); kf[6]  = to_bf(c1.z); kf[7]  = to_bf(c1.w);
        kf[8] = to_bf(c2.x); kf[9] = to_bf(c2.y); kf[10] = to_bf(c2.z); kf[11] = to_bf(c2.w);
        kf[12] = to_bf(c3.x); kf[13] = to_bf(c3.y); kf[14] = to_bf(c3.z); kf[15] = to_bf(c3.w);
        acc = wmma_bf16(qf[ks], kf, acc);
      }
#pragma unroll
      for (int r = 0; r < 8; ++r) {
        sS[(r + 8 * kh) * 1024 + jc + n] = to_bf(acc[r] * scale);
      }
    }
  }
  __syncthreads();

  // Phase 2: dual softmax + lam*(p1-p2). Each wave owns 2 rows.
  const float lamv = lam[layer];
#pragma unroll
  for (int rr = 0; rr < 2; ++rr) {
    const int row = wid * 2 + rr;
    const __bf16* srow = &sS[row * 1024];

    float m1 = -3.0e38f, m2 = -3.0e38f;
    for (int t = 0; t < 32; ++t) {
      const int j = lane + t * 32;
      const bool on = mask[b * SEQL + j] != 0;
      const float s1 = on ? (float)srow[j] : -1.0e9f;
      const float s2 = on ? (float)srow[(j + SEQL - 1) & (SEQL - 1)] : -1.0e9f;
      m1 = fmaxf(m1, s1);
      m2 = fmaxf(m2, s2);
    }
#pragma unroll
    for (int off = 16; off > 0; off >>= 1) {
      m1 = fmaxf(m1, __shfl_xor(m1, off));
      m2 = fmaxf(m2, __shfl_xor(m2, off));
    }

    float z1 = 0.f, z2 = 0.f;
    for (int t = 0; t < 32; ++t) {
      const int j = lane + t * 32;
      const bool on = mask[b * SEQL + j] != 0;
      const float s1 = on ? (float)srow[j] : -1.0e9f;
      const float s2 = on ? (float)srow[(j + SEQL - 1) & (SEQL - 1)] : -1.0e9f;
      z1 += expf(s1 - m1);
      z2 += expf(s2 - m2);
    }
#pragma unroll
    for (int off = 16; off > 0; off >>= 1) {
      z1 += __shfl_xor(z1, off);
      z2 += __shfl_xor(z2, off);
    }
    const float i1 = 1.f / z1;
    const float i2 = 1.f / z2;

    for (int t = 0; t < 32; ++t) {
      const int j = lane + t * 32;
      const bool on = mask[b * SEQL + j] != 0;
      const float s1 = on ? (float)srow[j] : -1.0e9f;
      const float s2 = on ? (float)srow[(j + SEQL - 1) & (SEQL - 1)] : -1.0e9f;
      const float d = lamv * (expf(s1 - m1) * i1 - expf(s2 - m2) * i2);
      sD[row * 1024 + j] = to_bf(d);
    }
  }
  __syncthreads();

  // Phase 3: o(16x64) = diff(16x1024) @ v(1024x64). Waves 0..3, one 16x16
  // column tile each, full-K accumulation.
  if (wid < 4) {
    const int c0 = wid * 16;
    const int n  = lane & 15;
    const int kh = (lane >> 4) & 1;
    v8f acc = v8zero();
    for (int kt = 0; kt < SEQL; kt += 32) {
      const v16bf af = lds_frag_a(sD, 1024, 0, kt, lane);
      v16bf vf;
#pragma unroll
      for (int pr = 0; pr < 8; ++pr) {
        const int j0 = kt + kh * 8 + 2 * (pr & 3) + (pr >> 2) * 16;  // sequence positions j0, j0+1
        const float* v0 =
            qkv + (size_t)(b * SEQL + j0) * rstride + 2 * BDIM + h * HDIM + c0 + n;
        vf[2 * pr]     = to_bf(v0[0]);
        vf[2 * pr + 1] = to_bf(v0[rstride]);
      }
      acc = wmma_bf16(af, vf, acc);
    }
#pragma unroll
    for (int r = 0; r < 8; ++r) {
      const int m = rowBase + r + 8 * kh;
      attn[(size_t)(b * SEQL + m) * BDIM + h * HDIM + c0 + n] = acc[r];
    }
  }
}

// ---------------------------------------------------------------------------
// Row kernels (one block of 256 threads per (b,l) row).
// ---------------------------------------------------------------------------
__global__ __launch_bounds__(256)
void prep_kernel(const float* __restrict__ x, const float* __restrict__ cw,
                 const float* __restrict__ cb, const float* __restrict__ lng,
                 const float* __restrict__ lnb, float* __restrict__ x_pos,
                 float* __restrict__ delta) {
  const int row = blockIdx.x;
  const int l   = row & (SEQL - 1);
  const int tid = threadIdx.x;
  const float* xr = x + (size_t)row * BDIM;
  const float* xp = (l > 0) ? xr - BDIM : nullptr;
  const float* xn = (l < SEQL - 1) ? xr + BDIM : nullptr;

  float c[4];
#pragma unroll
  for (int j = 0; j < 4; ++j) {
    const int d = tid + j * 256;
    const float a0 = xp ? xp[d] : 0.f;
    const float a2 = xn ? xn[d] : 0.f;
    c[j] = cw[d * 3 + 0] * a0 + cw[d * 3 + 1] * xr[d] + cw[d * 3 + 2] * a2 + cb[d];
  }
  float part = c[0] + c[1] + c[2] + c[3];
  const float mean = block_sum(part) * (1.f / BDIM);
  part = 0.f;
#pragma unroll
  for (int j = 0; j < 4; ++j) {
    const float t = c[j] - mean;
    part += t * t;
  }
  const float var = block_sum(part) * (1.f / BDIM);
  const float inv = rsqrtf(var + 1e-5f);
#pragma unroll
  for (int j = 0; j < 4; ++j) {
    const int d = tid + j * 256;
    delta[(size_t)row * BDIM + d] = (c[j] - mean) * inv * lng[d] + lnb[d];
  }

#pragma unroll
  for (int j = 0; j < 2; ++j) {
    const int i = tid + j * 256;  // pair index 0..511
    const float invf = powf(10000.f, -2.f * (float)i / (float)BDIM);
    const float fr = (float)l * invf;
    const float cs = cosf(fr), sn = sinf(fr);
    const float x1 = xr[2 * i], x2 = xr[2 * i + 1];
    x_pos[(size_t)row * BDIM + 2 * i]     = x1 * cs - x2 * sn;
    x_pos[(size_t)row * BDIM + 2 * i + 1] = x1 * sn + x2 * cs;
  }
}

__global__ __launch_bounds__(256)
void deltax_gi_kernel(const float* __restrict__ x, const float* __restrict__ gate,
                      const float* __restrict__ delta,
                      const float* __restrict__ alpha_p,
                      const float* __restrict__ x_pos,
                      const float* __restrict__ g2, const float* __restrict__ b2,
                      float* __restrict__ dxout, float* __restrict__ gi) {
  __shared__ float sdx[BDIM];
  const int row = blockIdx.x;
  const int tid = threadIdx.x;
  const float alpha = alpha_p[0];
  const size_t ro = (size_t)row * BDIM;
#pragma unroll
  for (int j = 0; j < 4; ++j) {
    const int d = tid + j * 256;
    const float dx = x[ro + d] + alpha * gate[ro + d] * delta[ro + d];
    sdx[d] = dx;
    dxout[ro + d] = dx;
  }
  __syncthreads();

  float v[8];
  float part = 0.f;
#pragma unroll
  for (int j = 0; j < 8; ++j) {
    const int e = tid + j * 256;
    const float t = (e < BDIM) ? x_pos[ro + e] : sdx[e - BDIM];
    v[j] = t;
    part += t;
  }
  const float mean = block_sum(part) * (1.f / (2 * BDIM));
  part = 0.f;
#pragma unroll
  for (int j = 0; j < 8; ++j) {
    const float t = v[j] - mean;
    part += t * t;
  }
  const float var = block_sum(part) * (1.f / (2 * BDIM));
  const float inv = rsqrtf(var + 1e-5f);
#pragma unroll
  for (int j = 0; j < 8; ++j) {
    const int e = tid + j * 256;
    gi[(size_t)row * (2 * BDIM) + e] = (v[j] - mean) * inv * g2[e] + b2[e];
  }
}

__global__ __launch_bounds__(256)
void hgate_kernel(const float* __restrict__ tbuf, const float* __restrict__ w2,
                  const float* __restrict__ b2, const float* __restrict__ dx,
                  const float* __restrict__ x_pos, const float* __restrict__ pg,
                  const float* __restrict__ pb, float* __restrict__ h) {
  const int row = blockIdx.x;
  const int tid = threadIdx.x;
  const size_t ro = (size_t)row * BDIM;
  float part = 0.f;
#pragma unroll
  for (int j = 0; j < 4; ++j) {
    const int d = tid + j * 256;
    part += tbuf[ro + d] * w2[d];
  }
  const float dot = block_sum(part);
  const float g = 1.f / (1.f + expf(-(dot + b2[0])));

  float v[4];
  part = 0.f;
#pragma unroll
  for (int j = 0; j < 4; ++j) {
    const int d = tid + j * 256;
    const float xv = x_pos[ro + d];
    const float hv = g * dx[ro + d] + (1.f - g) * xv + xv;
    v[j] = hv;
    part += hv;
  }
  const float mean = block_sum(part) * (1.f / BDIM);
  part = 0.f;
#pragma unroll
  for (int j = 0; j < 4; ++j) {
    const float t = v[j] - mean;
    part += t * t;
  }
  const float var = block_sum(part) * (1.f / BDIM);
  const float inv = rsqrtf(var + 1e-6f);
#pragma unroll
  for (int j = 0; j < 4; ++j) {
    const int d = tid + j * 256;
    h[ro + d] = (v[j] - mean) * inv * pg[d] + pb[d];
  }
}

__global__ __launch_bounds__(256)
void postattn_kernel(const float* __restrict__ hin, const float* __restrict__ oproj,
                     const float* __restrict__ g1, const float* __restrict__ b1,
                     const float* __restrict__ fg, const float* __restrict__ fb,
                     float* __restrict__ hout, float* __restrict__ ffn_in) {
  const int row = blockIdx.x;
  const int tid = threadIdx.x;
  const size_t ro = (size_t)row * BDIM;

  float v[4];
  float part = 0.f;
#pragma unroll
  for (int j = 0; j < 4; ++j) {
    const int d = tid + j * 256;
    const float t = hin[ro + d] + oproj[ro + d];
    v[j] = t;
    part += t;
  }
  float mean = block_sum(part) * (1.f / BDIM);
  part = 0.f;
#pragma unroll
  for (int j = 0; j < 4; ++j) {
    const float t = v[j] - mean;
    part += t * t;
  }
  float var = block_sum(part) * (1.f / BDIM);
  float inv = rsqrtf(var + 1e-6f);

  float hn[4];
  part = 0.f;
#pragma unroll
  for (int j = 0; j < 4; ++j) {
    const int d = tid + j * 256;
    hn[j] = (v[j] - mean) * inv * g1[d] + b1[d];
    hout[ro + d] = hn[j];
    part += hn[j];
  }
  mean = block_sum(part) * (1.f / BDIM);
  part = 0.f;
#pragma unroll
  for (int j = 0; j < 4; ++j) {
    const float t = hn[j] - mean;
    part += t * t;
  }
  var = block_sum(part) * (1.f / BDIM);
  inv = rsqrtf(var + 1e-6f);
#pragma unroll
  for (int j = 0; j < 4; ++j) {
    const int d = tid + j * 256;
    ffn_in[ro + d] = (hn[j] - mean) * inv * fg[d] + fb[d];
  }
}

__global__ __launch_bounds__(256)
void postffn_kernel(const float* __restrict__ hin, const float* __restrict__ ff0,
                    const float* __restrict__ g2, const float* __restrict__ b2,
                    float* __restrict__ hout) {
  const int row = blockIdx.x;
  const int tid = threadIdx.x;
  const size_t ro = (size_t)row * BDIM;

  float v[4];
  float part = 0.f;
#pragma unroll
  for (int j = 0; j < 4; ++j) {
    const int d = tid + j * 256;
    const float hv = hin[ro + d];
    const float t = hv + (ff0[ro + d] + hv);
    v[j] = t;
    part += t;
  }
  const float mean = block_sum(part) * (1.f / BDIM);
  part = 0.f;
#pragma unroll
  for (int j = 0; j < 4; ++j) {
    const float t = v[j] - mean;
    part += t * t;
  }
  const float var = block_sum(part) * (1.f / BDIM);
  const float inv = rsqrtf(var + 1e-6f);
#pragma unroll
  for (int j = 0; j < 4; ++j) {
    const int d = tid + j * 256;
    hout[ro + d] = (v[j] - mean) * inv * g2[d] + b2[d];
  }
}

__global__ __launch_bounds__(256)
void final_kernel(const float* __restrict__ x_pos, const float* __restrict__ h,
                  float* __restrict__ out, int n) {
  const int i = blockIdx.x * 256 + threadIdx.x;
  if (i < n) out[i] = x_pos[i] + 0.5f * h[i];
}

// ---------------------------------------------------------------------------
// Host launch.
// ---------------------------------------------------------------------------
enum {
  IN_X = 0, IN_MASK, IN_DCONV_W, IN_DCONV_B, IN_DLN_G, IN_DLN_B,
  IN_DGATE_W, IN_DGATE_B, IN_DALPHA, IN_CGLN_G, IN_CGLN_B,
  IN_CG_W1, IN_CG_B1, IN_CG_W2, IN_CG_B2, IN_PRE_G, IN_PRE_B,
  IN_QKV_W, IN_QKV_B, IN_OUT_W, IN_OUT_B, IN_LAM,
  IN_N1_G, IN_N1_B, IN_N2_G, IN_N2_B, IN_FFNLN_G, IN_FFNLN_B,
  IN_FFN_W1, IN_FFN_B1, IN_FFN_W2, IN_FFN_B2
};

extern "C" void kernel_launch(void* const* d_in, const int* in_sizes, int n_in,
                              void* d_out, int out_size, void* d_ws, size_t ws_size,
                              hipStream_t stream) {
  (void)in_sizes; (void)n_in; (void)out_size; (void)ws_size;

  const float* x       = (const float*)d_in[IN_X];
  const int*   mask    = (const int*)  d_in[IN_MASK];
  const float* dconv_w = (const float*)d_in[IN_DCONV_W];
  const float* dconv_b = (const float*)d_in[IN_DCONV_B];
  const float* dln_g   = (const float*)d_in[IN_DLN_G];
  const float* dln_b   = (const float*)d_in[IN_DLN_B];
  const float* dgate_w = (const float*)d_in[IN_DGATE_W];
  const float* dgate_b = (const float*)d_in[IN_DGATE_B];
  const float* dalpha  = (const float*)d_in[IN_DALPHA];
  const float* cgln_g  = (const float*)d_in[IN_CGLN_G];
  const float* cgln_b  = (const float*)d_in[IN_CGLN_B];
  const float* cg_w1   = (const float*)d_in[IN_CG_W1];
  const float* cg_b1   = (const float*)d_in[IN_CG_B1];
  const float* cg_w2   = (const float*)d_in[IN_CG_W2];
  const float* cg_b2   = (const float*)d_in[IN_CG_B2];
  const float* pre_g   = (const float*)d_in[IN_PRE_G];
  const float* pre_b   = (const float*)d_in[IN_PRE_B];
  const float* qkv_w   = (const float*)d_in[IN_QKV_W];
  const float* qkv_b   = (const float*)d_in[IN_QKV_B];
  const float* out_w   = (const float*)d_in[IN_OUT_W];
  const float* out_b   = (const float*)d_in[IN_OUT_B];
  const float* lam     = (const float*)d_in[IN_LAM];
  const float* n1_g    = (const float*)d_in[IN_N1_G];
  const float* n1_b    = (const float*)d_in[IN_N1_B];
  const float* n2_g    = (const float*)d_in[IN_N2_G];
  const float* n2_b    = (const float*)d_in[IN_N2_B];
  const float* fln_g   = (const float*)d_in[IN_FFNLN_G];
  const float* fln_b   = (const float*)d_in[IN_FFNLN_B];
  const float* ffn_w1  = (const float*)d_in[IN_FFN_W1];
  const float* ffn_b1  = (const float*)d_in[IN_FFN_B1];
  const float* ffn_w2  = (const float*)d_in[IN_FFN_W2];
  const float* ffn_b2  = (const float*)d_in[IN_FFN_B2];

  const int M = NBAT * SEQL;              // 4096 rows
  const size_t n1 = (size_t)M * BDIM;     // B*L*D elements

  float* ws    = (float*)d_ws;
  float* x_pos = ws + 0 * n1;
  float* delta = ws + 1 * n1;   // reused as oproj
  float* gate  = ws + 2 * n1;   // reused as ffn_in
  float* dx    = ws + 3 * n1;   // reused as attn output
  float* gi    = ws + 4 * n1;   // 2*n1; region 4..7 reused as qkv after h
  float* tbuf  = ws + 6 * n1;   // dead after hgate
  float* hbuf  = ws + 7 * n1;
  float* qkvb  = ws + 4 * n1;   // 3*n1 (overlaps dead gi/tbuf)
  float* inter = ws + 8 * n1;   // 4*n1
  float* ff0   = ws + 12 * n1;

  // bf16 transposed weights live after the f32 region (27M bf16 elements).
  __bf16* wsb   = (__bf16*)(ws + 16 * n1);
  const size_t MEG = 1024 * 1024;
  __bf16* bgate = wsb + 0;
  __bf16* bcg1  = wsb + 1 * MEG;                // 2M
  __bf16* bqkv  = wsb + 3 * MEG;                // 2 x 3M
  __bf16* bout  = wsb + 9 * MEG;                // 2 x 1M
  __bf16* bffn1 = wsb + 11 * MEG;               // 2 x 4M
  __bf16* bffn2 = wsb + 19 * MEG;               // 2 x 4M

  float* attn   = dx;
  float* oproj  = delta;
  float* ffn_in = gate;

  const dim3 blk(256);
  const dim3 tblk(32, 8);

  // ---- weight prep: f32 KxN -> bf16 NxK ----
  wtrans_kernel<<<dim3(BDIM / 32, BDIM / 32), tblk, 0, stream>>>(
      dgate_w, bgate, BDIM, BDIM);
  wtrans_kernel<<<dim3(BDIM / 32, (2 * BDIM) / 32), tblk, 0, stream>>>(
      cg_w1, bcg1, 2 * BDIM, BDIM);
  for (int i = 0; i < 2; ++i) {
    wtrans_kernel<<<dim3((3 * BDIM) / 32, BDIM / 32), tblk, 0, stream>>>(
        qkv_w + (size_t)i * BDIM * 3 * BDIM, bqkv + (size_t)i * 3 * MEG,
        BDIM, 3 * BDIM);
    wtrans_kernel<<<dim3(BDIM / 32, BDIM / 32), tblk, 0, stream>>>(
        out_w + (size_t)i * BDIM * BDIM, bout + (size_t)i * MEG, BDIM, BDIM);
    wtrans_kernel<<<dim3(DFFN / 32, BDIM / 32), tblk, 0, stream>>>(
        ffn_w1 + (size_t)i * BDIM * DFFN, bffn1 + (size_t)i * 4 * MEG,
        BDIM, DFFN);
    wtrans_kernel<<<dim3(BDIM / 32, DFFN / 32), tblk, 0, stream>>>(
        ffn_w2 + (size_t)i * DFFN * BDIM, bffn2 + (size_t)i * 4 * MEG,
        DFFN, BDIM);
  }

  // ---- forward pass ----
  prep_kernel<<<dim3(M), blk, 0, stream>>>(x, dconv_w, dconv_b, dln_g, dln_b,
                                           x_pos, delta);
  gemm_bf16<ACT_SIGMOID><<<dim3(BDIM / 128, M / 128), blk, 0, stream>>>(
      x, bgate, dgate_b, gate, M, BDIM, BDIM);
  deltax_gi_kernel<<<dim3(M), blk, 0, stream>>>(x, gate, delta, dalpha, x_pos,
                                                cgln_g, cgln_b, dx, gi);
  gemm_bf16<ACT_TANH><<<dim3(BDIM / 128, M / 128), blk, 0, stream>>>(
      gi, bcg1, cg_b1, tbuf, M, BDIM, 2 * BDIM);
  hgate_kernel<<<dim3(M), blk, 0, stream>>>(tbuf, cg_w2, cg_b2, dx, x_pos,
                                            pre_g, pre_b, hbuf);

  for (int i = 0; i < 2; ++i) {
    gemm_bf16<ACT_NONE><<<dim3((3 * BDIM) / 128, M / 128), blk, 0, stream>>>(
        hbuf, bqkv + (size_t)i * 3 * MEG, qkv_b + (size_t)i * 3 * BDIM,
        qkvb, M, 3 * BDIM, BDIM);
    attn_kernel<<<dim3(SEQL / 16, NBAT * NHEAD), blk, 0, stream>>>(
        qkvb, mask, lam, i, attn);
    gemm_bf16<ACT_NONE><<<dim3(BDIM / 128, M / 128), blk, 0, stream>>>(
        attn, bout + (size_t)i * MEG, out_b + (size_t)i * BDIM,
        oproj, M, BDIM, BDIM);
    postattn_kernel<<<dim3(M), blk, 0, stream>>>(
        hbuf, oproj, n1_g + (size_t)i * BDIM, n1_b + (size_t)i * BDIM,
        fln_g + (size_t)i * BDIM, fln_b + (size_t)i * BDIM, hbuf, ffn_in);
    gemm_bf16<ACT_GELU><<<dim3(DFFN / 128, M / 128), blk, 0, stream>>>(
        ffn_in, bffn1 + (size_t)i * 4 * MEG, ffn_b1 + (size_t)i * DFFN,
        inter, M, DFFN, BDIM);
    gemm_bf16<ACT_NONE><<<dim3(BDIM / 128, M / 128), blk, 0, stream>>>(
        inter, bffn2 + (size_t)i * 4 * MEG, ffn_b2 + (size_t)i * BDIM,
        ff0, M, BDIM, DFFN);
    postffn_kernel<<<dim3(M), blk, 0, stream>>>(
        hbuf, ff0, n2_g + (size_t)i * BDIM, n2_b + (size_t)i * BDIM, hbuf);
  }

  final_kernel<<<dim3((int)(n1 / 256)), blk, 0, stream>>>(
      x_pos, hbuf, (float*)d_out, (int)n1);
}
